// NodeModel_73959336837503
// MI455X (gfx1250) — compile-verified, matrix operand order checked by reading, not run on hardware
//
#include <hip/hip_runtime.h>

typedef __attribute__((ext_vector_type(16))) _Float16 v16h;
typedef __attribute__((ext_vector_type(8)))  float    v8f;

namespace {
constexpr int kN = 50000;
constexpr int kE = 1600000;
constexpr int kDIn = 32, kDEdge = 32, kDH = 64, kDU = 16, kDOut = 32;
constexpr int kEdgeFeat = kDIn + kDEdge;       // 64
constexpr int kNodeFeat = kDIn + kDH + kDU;    // 112
constexpr int kNodeFeatPad = 128;

// workspace layout (bytes); all weight blocks 32B-aligned for v16h loads
constexpr size_t kOffSummed = 0;                                             // float[N*64]
constexpr size_t kOffCounts = kOffSummed + size_t(kN) * kDH * sizeof(float); // float[N]
constexpr size_t kOffW1a = (kOffCounts + size_t(kN) * sizeof(float) + 255) & ~size_t(255);
constexpr size_t kOffW1b = kOffW1a + size_t(kEdgeFeat) * kDH * sizeof(_Float16);      // 8192 B
constexpr size_t kOffW2a = kOffW1b + size_t(kDH) * kDH * sizeof(_Float16);            // 8192 B
constexpr size_t kOffW2b = kOffW2a + size_t(kNodeFeatPad) * kDH * sizeof(_Float16);   // 16384 B
} // namespace

// ---- WMMA fragment index helper (CDNA5 16-bit A/B layout, wave32) ----
// Half-element i = 2v+b of lane L maps to k = 16*(v>>2) + 8*(L>>4) + 2*(v&3) + b.
// A fragment: row m = L&15.  B fragment: output column n = ntile*16 + (L&15).
__device__ __forceinline__ int frag_k(int lane, int i) {
    int g = lane >> 4;
    int v = i >> 1, b = i & 1;
    return 16 * (v >> 2) + 8 * g + 2 * (v & 3) + b;
}

// ---------------- prep kernels ----------------
__global__ void zero_f32(float* __restrict__ p, int n) {
    int i = blockIdx.x * blockDim.x + threadIdx.x;
    if (i < n) p[i] = 0.0f;
}

// Pre-swizzle a row-major [Kreal x ldn] f32 weight matrix into fragment-major f16:
// out[((kc*nn + nt)*32 + lane)*16 + i] = W[(kc*32 + frag_k(lane,i)) * ldn + nt*16 + (lane&15)]
// so each lane's 16 halfs of a B fragment are 32 contiguous, 32B-aligned bytes.
__device__ __forceinline__ void swizzle_one(const float* __restrict__ W, _Float16* __restrict__ out,
                                            int o, int Kreal, int ldn, int nn) {
    const int f    = o >> 9;         // /512 elements per fragment
    const int lane = (o >> 4) & 31;
    const int i    = o & 15;
    const int kc = f / nn, nt = f % nn;
    const int k = kc * 32 + frag_k(lane, i);
    const int n = nt * 16 + (lane & 15);
    const float v = (k < Kreal) ? W[k * ldn + n] : 0.0f;
    out[o] = (_Float16)v;
}

__global__ void cvt_weights(const float* __restrict__ W1a, const float* __restrict__ W1b,
                            const float* __restrict__ W2a, const float* __restrict__ W2b,
                            _Float16* __restrict__ w1a, _Float16* __restrict__ w1b,
                            _Float16* __restrict__ w2a, _Float16* __restrict__ w2b) {
    int i = blockIdx.x * blockDim.x + threadIdx.x;
    if (i < kEdgeFeat * kDH)     swizzle_one(W1a, w1a, i, kEdgeFeat, kDH, 4);  // 2 kc x 4 nt
    if (i < kDH * kDH)           swizzle_one(W1b, w1b, i, kDH, kDH, 4);        // 2 kc x 4 nt
    if (i < kNodeFeatPad * kDH)  swizzle_one(W2a, w2a, i, kNodeFeat, kDH, 4);  // 4 kc x 4 nt (K-pad)
    if (i < kDH * kDOut)         swizzle_one(W2b, w2b, i, kDH, kDOut, 2);      // 2 kc x 2 nt
}

// ---------------- edge MLP + scatter-add ----------------
// 1 wave == 16 edges.  8 waves/block, grid = E/16/8 = 12500 (exact).
__global__ __launch_bounds__(256) void edge_mlp(
    const float* __restrict__ x, const int* __restrict__ ei, const float* __restrict__ ea,
    const v16h* __restrict__ w1a, const float* __restrict__ b1a,
    const v16h* __restrict__ w1b, const float* __restrict__ b1b,
    float* __restrict__ summed, float* __restrict__ counts) {
    __shared__ float stage[8][16 * kDH];  // 32 KB, per-wave slabs

    const int lane = threadIdx.x & 31;
    const int wib  = threadIdx.x >> 5;
    const int tile = blockIdx.x * 8 + wib;
    const int e0   = tile * 16;
    const int g    = lane >> 4;
    const int m    = lane & 15;

    const int eRow = e0 + m;                 // edge this lane loads features for
    const int src  = ei[kE + eRow];          // col = edge_index[1][e] (gather)

    // A fragments for layer 1: features = [x[src] (k 0..31) || edge_attr (k 32..63)]
    // per lane these are two contiguous 8-float runs -> vectorized b128 loads
    v16h a0, a1;
#pragma unroll
    for (int i = 0; i < 16; ++i) {
        int k = frag_k(lane, i);             // 0..31
        a0[i] = (_Float16)x[src * kDIn + k];
        a1[i] = (_Float16)ea[eRow * kDEdge + k];
    }

    // layer 1: [16x64] @ [64x64] -> relu -> stage
#pragma unroll
    for (int nt = 0; nt < 4; ++nt) {
        v8f acc = {};
        v16h b0 = w1a[(0 * 4 + nt) * 32 + lane];   // pre-swizzled fragment, 32B/lane
        acc = __builtin_amdgcn_wmma_f32_16x16x32_f16(false, a0, false, b0, (short)0, acc, false, false);
        v16h b1 = w1a[(1 * 4 + nt) * 32 + lane];
        acc = __builtin_amdgcn_wmma_f32_16x16x32_f16(false, a1, false, b1, (short)0, acc, false, false);
        const int c = nt * 16 + m;           // D: lane -> column, vgpr j -> row (j + 8*g)
#pragma unroll
        for (int j = 0; j < 8; ++j) {
            float v = acc[j] + b1a[c];
            v = v > 0.0f ? v : 0.0f;
            stage[wib][(j + 8 * g) * kDH + c] = v;
        }
    }
    __syncthreads();

    // re-fragment for layer 2 (contiguous 8-float runs -> ds_load vectorized)
    v16h c0, c1;
#pragma unroll
    for (int i = 0; i < 16; ++i) {
        int k = frag_k(lane, i);
        c0[i] = (_Float16)stage[wib][m * kDH + k];
        c1[i] = (_Float16)stage[wib][m * kDH + 32 + k];
    }

    // layer 2: [16x64] @ [64x64] -> relu -> atomic scatter into summed[dst]
#pragma unroll
    for (int nt = 0; nt < 4; ++nt) {
        v8f acc = {};
        v16h b0 = w1b[(0 * 4 + nt) * 32 + lane];
        acc = __builtin_amdgcn_wmma_f32_16x16x32_f16(false, c0, false, b0, (short)0, acc, false, false);
        v16h b1 = w1b[(1 * 4 + nt) * 32 + lane];
        acc = __builtin_amdgcn_wmma_f32_16x16x32_f16(false, c1, false, b1, (short)0, acc, false, false);
        const int c = nt * 16 + m;
#pragma unroll
        for (int j = 0; j < 8; ++j) {
            float v = acc[j] + b1b[c];
            v = v > 0.0f ? v : 0.0f;
            const int e   = e0 + j + 8 * g;
            const int dst = ei[e];           // row = edge_index[0][e]
            atomicAdd(&summed[dst * kDH + c], v);
        }
    }
    if (lane < 16) atomicAdd(&counts[ei[e0 + lane]], 1.0f);
}

// ---------------- node MLP ----------------
// 1 wave == 16 nodes.  5 waves/block x 625 blocks = 3125 tiles = N/16 (exact).
__global__ __launch_bounds__(160) void node_mlp(
    const float* __restrict__ x, const float* __restrict__ u, const int* __restrict__ batch,
    const v16h* __restrict__ w2a, const float* __restrict__ b2a,
    const v16h* __restrict__ w2b, const float* __restrict__ b2b,
    const float* __restrict__ summed, const float* __restrict__ counts,
    float* __restrict__ out) {
    __shared__ float stage[5][16 * kDH];  // 20 KB

    const int lane = threadIdx.x & 31;
    const int wib  = threadIdx.x >> 5;
    const int tile = blockIdx.x * 5 + wib;
    const int n0   = tile * 16;
    const int g    = lane >> 4;
    const int m    = lane & 15;

    const int   node = n0 + m;
    const float invc = 1.0f / fmaxf(counts[node], 1.0f);
    const int   ub   = batch[node];

    // A = [x || agg || u[batch] || zero-pad] : K = 128 in 4 chunks of 32
    v16h a[4];
#pragma unroll
    for (int c = 0; c < 4; ++c) {
#pragma unroll
        for (int i = 0; i < 16; ++i) {
            const int k = c * 32 + frag_k(lane, i);
            float v;
            if (k < kDIn)              v = x[node * kDIn + k];
            else if (k < kDIn + kDH)   v = summed[node * kDH + (k - kDIn)] * invc;
            else if (k < kNodeFeat)    v = u[ub * kDU + (k - kDIn - kDH)];
            else                       v = 0.0f;
            a[c][i] = (_Float16)v;
        }
    }

    // layer 1: [16x128] @ [128x64] -> relu -> stage
#pragma unroll
    for (int nt = 0; nt < 4; ++nt) {
        v8f acc = {};
#pragma unroll
        for (int c = 0; c < 4; ++c) {
            v16h b = w2a[(c * 4 + nt) * 32 + lane];
            acc = __builtin_amdgcn_wmma_f32_16x16x32_f16(false, a[c], false, b, (short)0, acc, false, false);
        }
        const int cc = nt * 16 + m;
#pragma unroll
        for (int j = 0; j < 8; ++j) {
            float v = acc[j] + b2a[cc];
            v = v > 0.0f ? v : 0.0f;
            stage[wib][(j + 8 * g) * kDH + cc] = v;
        }
    }
    __syncthreads();

    v16h h0, h1;
#pragma unroll
    for (int i = 0; i < 16; ++i) {
        int k = frag_k(lane, i);
        h0[i] = (_Float16)stage[wib][m * kDH + k];
        h1[i] = (_Float16)stage[wib][m * kDH + 32 + k];
    }

    // layer 2: [16x64] @ [64x32] -> out
#pragma unroll
    for (int nt = 0; nt < 2; ++nt) {
        v8f acc = {};
        v16h b0 = w2b[(0 * 2 + nt) * 32 + lane];
        acc = __builtin_amdgcn_wmma_f32_16x16x32_f16(false, h0, false, b0, (short)0, acc, false, false);
        v16h b1 = w2b[(1 * 2 + nt) * 32 + lane];
        acc = __builtin_amdgcn_wmma_f32_16x16x32_f16(false, h1, false, b1, (short)0, acc, false, false);
        const int cc = nt * 16 + m;
#pragma unroll
        for (int j = 0; j < 8; ++j) {
            out[(n0 + j + 8 * g) * kDOut + cc] = acc[j] + b2b[cc];
        }
    }
}

extern "C" void kernel_launch(void* const* d_in, const int* in_sizes, int n_in,
                              void* d_out, int out_size, void* d_ws, size_t ws_size,
                              hipStream_t stream) {
    const float* x    = (const float*)d_in[0];
    const int*   ei   = (const int*)d_in[1];
    const float* ea   = (const float*)d_in[2];
    const float* u    = (const float*)d_in[3];
    const int*   batch= (const int*)d_in[4];
    const float* W1a  = (const float*)d_in[5];
    const float* b1a  = (const float*)d_in[6];
    const float* W1b  = (const float*)d_in[7];
    const float* b1b  = (const float*)d_in[8];
    const float* W2a  = (const float*)d_in[9];
    const float* b2a  = (const float*)d_in[10];
    const float* W2b  = (const float*)d_in[11];
    const float* b2b  = (const float*)d_in[12];
    float* out = (float*)d_out;

    char* ws = (char*)d_ws;
    float*    summed = (float*)(ws + kOffSummed);
    float*    counts = (float*)(ws + kOffCounts);
    _Float16* w1a    = (_Float16*)(ws + kOffW1a);
    _Float16* w1b    = (_Float16*)(ws + kOffW1b);
    _Float16* w2a    = (_Float16*)(ws + kOffW2a);
    _Float16* w2b    = (_Float16*)(ws + kOffW2b);

    // 1) zero accumulators (summed and counts are contiguous)
    {
        const int n = kN * (kDH + 1);
        zero_f32<<<(n + 255) / 256, 256, 0, stream>>>(summed, n);
    }
    // 2) weights -> f16, pre-swizzled into WMMA B-fragment layout
    {
        const int n = kNodeFeatPad * kDH;  // largest = 8192
        cvt_weights<<<(n + 255) / 256, 256, 0, stream>>>(W1a, W1b, W2a, W2b, w1a, w1b, w2a, w2b);
    }
    // 3) edge MLP + scatter: E/16 = 100000 wave-tiles, 8 waves/block
    edge_mlp<<<kE / 16 / 8, 256, 0, stream>>>(x, ei, ea,
        (const v16h*)w1a, b1a, (const v16h*)w1b, b1b, summed, counts);
    // 4) node MLP: N/16 = 3125 wave-tiles, 5 waves/block -> 625 blocks exact
    node_mlp<<<kN / 16 / 5, 160, 0, stream>>>(x, u, batch,
        (const v16h*)w2a, b2a, (const v16h*)w2b, b2b, summed, counts, out);
}
